// QuantizedLinear_44427141710301
// MI455X (gfx1250) — matrix operand order repaired and checked by reading, not compiled, not run on hardware
//
#include <hip/hip_runtime.h>
#include <hip/hip_bf16.h>
#include <stdint.h>

// CDNA5 wave32 WMMA types
typedef __attribute__((ext_vector_type(16))) __bf16 v16bf;
typedef __attribute__((ext_vector_type(8)))  float  v8f;
// TDM descriptor groups (6-arg builtin: g0, g1, g2, g3, g4, cpol)
typedef __attribute__((ext_vector_type(4))) unsigned int ui32x4;
typedef __attribute__((ext_vector_type(8))) int          i32x8;
typedef __attribute__((ext_vector_type(4))) int          i32x4;

#define BM 128
#define BN 128
#define BK 32
#define LPAD 8   // +8 halfs -> row stride 40 halfs = 80B: 16B-aligned, conflict-free columns

__device__ __forceinline__ uint16_t f32_to_bf16_rne(float x) {
  uint32_t u = __float_as_uint(x);
  return (uint16_t)((u + 0x7FFFu + ((u >> 16) & 1u)) >> 16);
}

// Truncating two-term split: x ~= hi + lo (bf16 each, ~16.5 mantissa bits total).
__device__ __forceinline__ void split_bf16x2(float x, uint16_t& hi, uint16_t& lo) {
  const uint32_t u = __float_as_uint(x);
  hi = (uint16_t)(u >> 16);
  const float hf = __uint_as_float(u & 0xFFFF0000u);
  const float l  = x - hf;
  lo = (uint16_t)(__float_as_uint(l) >> 16);
}

// WMMA 16-bit A-fragment column permutation: swap bits 3 and 4 of k.
__device__ __forceinline__ int aperm(int k) {
  return (k & 7) | ((k & 8) << 1) | ((k & 16) >> 1);
}

// LDS byte address of a __shared__ pointer (flat LDS aperture: offset in ADDR[31:0]).
__device__ __forceinline__ uint32_t lds_addr_of(const void* p) {
  return (uint32_t)(uintptr_t)p;
}

// One block per weight row: scale = mean(|w|), Q = bf16(round(w/scale)) (exact: small ints).
__global__ __launch_bounds__(256) void quant_kernel(const float* __restrict__ W,
                                                    uint16_t* __restrict__ Q,
                                                    float* __restrict__ scales,
                                                    int K) {
  const int row = blockIdx.x;
  const float* w = W + (size_t)row * K;

  float s = 0.f;
  for (int i = threadIdx.x * 4; i < K; i += blockDim.x * 4) {
    float4 v = *(const float4*)(w + i);
    s += fabsf(v.x) + fabsf(v.y) + fabsf(v.z) + fabsf(v.w);
  }
  for (int o = 16; o > 0; o >>= 1) s += __shfl_down(s, o, 32);

  __shared__ float red[8];
  const int lane = threadIdx.x & 31, wave = threadIdx.x >> 5;
  if (lane == 0) red[wave] = s;
  __syncthreads();
  if (threadIdx.x == 0) {
    float t = 0.f;
    for (int i = 0; i < 8; ++i) t += red[i];
    red[0] = t;
  }
  __syncthreads();

  const float scale = red[0] / (float)K;
  const float inv   = 1.0f / scale;
  uint16_t* q = Q + (size_t)row * K;
  for (int i = threadIdx.x * 4; i < K; i += blockDim.x * 4) {
    float4 v = *(const float4*)(w + i);
    ushort4 o4;
    o4.x = f32_to_bf16_rne(rintf(v.x * inv));
    o4.y = f32_to_bf16_rne(rintf(v.y * inv));
    o4.z = f32_to_bf16_rne(rintf(v.z * inv));
    o4.w = f32_to_bf16_rne(rintf(v.w * inv));
    *(ushort4*)(q + i) = o4;
  }
  if (threadIdx.x == 0) scales[row] = scale;
}

// out[T,N] = input[T,K] @ Q[N,K]^T * scale[N] + bias[N]
// Split-bf16 activations: acc += A_hi*B + A_lo*B via v_wmma_f32_16x16x32_bf16.
// Double-buffered pipeline: B tiles DMA'd by the Tensor Data Mover (TENSORcnt),
// A tiles global->VGPR loads issued before the WMMA block so latency hides under XDL.
__global__ __launch_bounds__(256) void gemm_kernel(const float* __restrict__ A,
                                                   const uint16_t* __restrict__ Q,
                                                   const float* __restrict__ scales,
                                                   const float* __restrict__ bias,
                                                   float* __restrict__ C,
                                                   int T, int N, int K) {
  // A planes stored in natural k order; fragment reads (k = 16h..16h+15) are contiguous b128s.
  __shared__ uint16_t sAhi[2][BM][BK + LPAD];  // 2*10240 B
  __shared__ uint16_t sAlo[2][BM][BK + LPAD];  // 2*10240 B
  __shared__ uint16_t sB  [2][BN][BK + LPAD];  // 2*10240 B (TDM fills 64B rows + 16B pad)

  const int tid  = threadIdx.x;
  const int lane = tid & 31;
  const int wave = tid >> 5;      // 8 waves
  const int wm   = wave & 3;      // 4 waves along M (32 rows each)
  const int wn   = wave >> 2;     // 2 waves along N (64 cols each)
  const int l16  = lane & 15;
  const int h    = lane >> 4;     // half-wave select (ISA 16-bit layouts)

  const int m0 = blockIdx.x * BM;
  const int n0 = blockIdx.y * BN;

  v8f acc[2][4] = {};             // 2 M-frags x 4 N-frags of 16x16 f32

  // cooperative A-tile load coordinates
  const int ar = tid >> 3;            // A: 32 rows/pass, 4 passes
  const int ac = (tid & 7) * 4;       //    8x float4 per row (4 consecutive k)
  const int ap = ac;                  //    natural k position (contiguous run of 4)

  // ---- TDM descriptor (constant parts) for the 128x32 bf16 B tile ----
  // group1: data_size=2B, pad_enable, pad_interval=16 DWORDs (one 64B row),
  // pad_amount=4 DWORDs (16B)  => LDS row stride 80B == sB row stride.
  const uint32_t g1w0 = (1u << 16)  |  // data_size = 1 (2 bytes)
                        (1u << 20)  |  // pad_enable
                        (3u << 22)  |  // pad_interval: 16 DWORDs
                        (3u << 25);    // pad_amount:   4 DWORDs
  const i32x8 g1 = {
      (int)g1w0,
      (int)((uint32_t)K << 16),                            // tensor_dim0[15:0]
      (int)(((uint32_t)K >> 16) | ((uint32_t)N << 16)),    // dim0[31:16] | dim1[15:0]
      (int)(((uint32_t)N >> 16) | ((uint32_t)BK << 16)),   // dim1[31:16] | tile_dim0=32
      (int)(uint32_t)BM,                                   // tile_dim1=128, tile_dim2=0
      (int)(uint32_t)K,                                    // tensor_dim0_stride[31:0]
      0, 0
  };
  const i32x4 gz4 = {0, 0, 0, 0};
  const i32x8 gz8 = {0, 0, 0, 0, 0, 0, 0, 0};
  const uint32_t ldsB[2] = { lds_addr_of(&sB[0][0][0]), lds_addr_of(&sB[1][0][0]) };
  const uint64_t qtile = (uint64_t)(uintptr_t)(Q + (size_t)n0 * K);  // byte addr of row n0

  // ================= prologue: stage tile k0=0 into buffer 0 =================
  if (wave == 0) {
    const ui32x4 g0 = {
        1u, ldsB[0], (uint32_t)qtile,
        (uint32_t)((qtile >> 32) & 0x01FFFFFFu) | (2u << 30)
    };
    __builtin_amdgcn_tensor_load_to_lds(g0, g1, gz4, gz4, gz8, 0);
  }
  {
    float4 va[4];
#pragma unroll
    for (int p = 0; p < 4; ++p)
      va[p] = *(const float4*)(A + (size_t)(m0 + ar + p * 32) * K + ac);
#pragma unroll
    for (int p = 0; p < 4; ++p) {
      ushort4 hi4, lo4;
      split_bf16x2(va[p].x, hi4.x, lo4.x);
      split_bf16x2(va[p].y, hi4.y, lo4.y);
      split_bf16x2(va[p].z, hi4.z, lo4.z);
      split_bf16x2(va[p].w, hi4.w, lo4.w);
      *(ushort4*)&sAhi[0][ar + p * 32][ap] = hi4;
      *(ushort4*)&sAlo[0][ar + p * 32][ap] = lo4;
    }
  }
  __builtin_amdgcn_s_wait_tensorcnt(0);
  __syncthreads();

  // ================= main pipelined loop: one barrier per k-step =============
  int cur = 0;
  for (int k0 = 0; k0 < K; k0 += BK) {
    const int  k1        = k0 + BK;
    const bool have_next = (k1 < K);
    const int  nb        = cur ^ 1;

    // ---- 1. issue next tile's loads (TDM for B, global b128s for A), no waits ----
    float4 va[4];
    if (have_next) {
      if (wave == 0) {
        const uint64_t ga = qtile + (size_t)k1 * sizeof(uint16_t);
        const ui32x4 g0 = {
            1u, ldsB[nb], (uint32_t)ga,
            (uint32_t)((ga >> 32) & 0x01FFFFFFu) | (2u << 30)
        };
        __builtin_amdgcn_tensor_load_to_lds(g0, g1, gz4, gz4, gz8, 0);
      }
#pragma unroll
      for (int p = 0; p < 4; ++p)
        va[p] = *(const float4*)(A + (size_t)(m0 + ar + p * 32) * K + (k1 + ac));
      if (k1 + BK < K) {  // L2 prefetch two tiles ahead (global_prefetch_b8)
        __builtin_prefetch(A + (size_t)(m0 + ar) * K + (k1 + BK + ac), 0, 1);
      }
    }

    // ---- 2. compute on current buffer (hides the loads issued above) ----
    v16bf bf[4];
#pragma unroll
    for (int fn = 0; fn < 4; ++fn) {
      union { v16bf v; uint4 q[2]; } U;
      const uint16_t* src = &sB[cur][wn * 64 + fn * 16 + l16][16 * h];
      U.q[0] = *(const uint4*)(src);
      U.q[1] = *(const uint4*)(src + 8);
      bf[fn] = U.v;
    }
    v16bf ahi[2], alo[2];
#pragma unroll
    for (int fm = 0; fm < 2; ++fm) {
      const int row = wm * 32 + fm * 16 + l16;
      union { v16bf v; uint4 q[2]; } Hh, Ll;
      const uint16_t* srcH = &sAhi[cur][row][16 * h];
      const uint16_t* srcL = &sAlo[cur][row][16 * h];
      Hh.q[0] = *(const uint4*)(srcH);
      Hh.q[1] = *(const uint4*)(srcH + 8);
      Ll.q[0] = *(const uint4*)(srcL);
      Ll.q[1] = *(const uint4*)(srcL + 8);
      ahi[fm] = Hh.v;
      alo[fm] = Ll.v;
    }
#pragma unroll
    for (int fm = 0; fm < 2; ++fm)
#pragma unroll
      for (int fn = 0; fn < 4; ++fn)
        acc[fm][fn] = __builtin_amdgcn_wmma_f32_16x16x32_bf16(
            false, ahi[fm], false, bf[fn], (short)0, acc[fm][fn], false, false);
#pragma unroll
    for (int fm = 0; fm < 2; ++fm)
#pragma unroll
      for (int fn = 0; fn < 4; ++fn)
        acc[fm][fn] = __builtin_amdgcn_wmma_f32_16x16x32_bf16(
            false, alo[fm], false, bf[fn], (short)0, acc[fm][fn], false, false);

    // ---- 3. convert the landed A data into the next buffer ----
    if (have_next) {
#pragma unroll
      for (int p = 0; p < 4; ++p) {
        ushort4 hi4, lo4;
        split_bf16x2(va[p].x, hi4.x, lo4.x);
        split_bf16x2(va[p].y, hi4.y, lo4.y);
        split_bf16x2(va[p].z, hi4.z, lo4.z);
        split_bf16x2(va[p].w, hi4.w, lo4.w);
        *(ushort4*)&sAhi[nb][ar + p * 32][ap] = hi4;
        *(ushort4*)&sAlo[nb][ar + p * 32][ap] = lo4;
      }
    }

    // ---- 4. single barrier: next-buffer writes complete, current-buffer reads done ----
    __builtin_amdgcn_s_wait_tensorcnt(0);
    __syncthreads();
    cur = nb;
  }

  // ---- epilogue: D layout -> lane col n=l16, VGPR v -> row v + 8*h; apply scale/bias ----
#pragma unroll
  for (int fn = 0; fn < 4; ++fn) {
    const int n = n0 + wn * 64 + fn * 16 + l16;
    const float s = scales[n];
    const float b = bias[n];
#pragma unroll
    for (int fm = 0; fm < 2; ++fm) {
      const int mrow = m0 + wm * 32 + fm * 16 + 8 * h;
#pragma unroll
      for (int v = 0; v < 8; ++v) {
        C[(size_t)(mrow + v) * N + n] = acc[fm][fn][v] * s + b;
      }
    }
  }
}

extern "C" void kernel_launch(void* const* d_in, const int* in_sizes, int n_in,
                              void* d_out, int out_size, void* d_ws, size_t ws_size,
                              hipStream_t stream) {
  const float* input  = (const float*)d_in[0];   // [T, K] fp32
  const float* weight = (const float*)d_in[1];   // [N, K] fp32
  const float* bias   = (const float*)d_in[2];   // [N]    fp32
  float* out = (float*)d_out;                    // [T, N] fp32

  const int N = in_sizes[2];                     // 11008
  const int K = in_sizes[1] / N;                 // 4096
  const int T = in_sizes[0] / K;                 // 8192

  // workspace: scales[N] (f32), then Q[N*K] (bf16) ~ 90.2 MB total
  float* scales = (float*)d_ws;
  const size_t scales_bytes = (((size_t)N * sizeof(float)) + 255) & ~(size_t)255;
  uint16_t* Q = (uint16_t*)((char*)d_ws + scales_bytes);

  quant_kernel<<<N, 256, 0, stream>>>(weight, Q, scales, K);

  dim3 grid(T / BM, N / BN);                     // 64 x 86 (shapes divide evenly)
  gemm_kernel<<<grid, 256, 0, stream>>>(input, Q, scales, bias, out, T, N, K);
}
